// ResBlock_3d_82471962018108
// MI455X (gfx1250) — matrix operand order repaired and checked by reading, not compiled, not run on hardware
//
#include <hip/hip_runtime.h>
#include <hip/hip_bf16.h>

// ---------------------------------------------------------------------------
// CDNA5 (gfx1250) deformable-conv ResBlock using V_WMMA_F32_16X16X32_F16.
//   xh / yh : channel-last f16  [B][T][H][W][C]  (C=64 contiguous, 128B rows)
//   wT      : f16 [64][1728], k = kv*64 + c
//   offf    : f32 [P][64] channel-last offset field (P = B*T*H*W)
// Workgroup = 4 waves = one (b,t,h) row (4 x 16-wide M tiles, N=64).
// Per-tap 64x64 f16 weight block staged in LDS (double-buffered, async copy).
// ---------------------------------------------------------------------------

typedef __attribute__((ext_vector_type(16))) _Float16 v16h;
typedef __attribute__((ext_vector_type(8)))  _Float16 v8h;
typedef __attribute__((ext_vector_type(8)))  float    v8f;

union AFrag { v16h v; v8h h[2]; };

#define C_IN  64
#define T_D   4
#define H_D   64
#define W_D   64
#define KVOL  27
#define KDIM  (C_IN * KVOL)   // 1728
#define OFFCH 54
#define LPITCH 72             // LDS row pitch in f16 (144B, conflict-free)

#if __has_builtin(__builtin_amdgcn_global_load_async_to_lds_b128)
#define USE_ASYNC 1
#else
#define USE_ASYNC 0
#endif

// builtin pointee/addr-space types:
//   param0: int __vector(4) addrspace(1)*   param1: int __vector(4) addrspace(3)*
typedef int v4i_ __attribute__((vector_size(16)));
typedef __attribute__((address_space(1))) v4i_ g_v4i;
typedef __attribute__((address_space(3))) v4i_ l_v4i;

__device__ __forceinline__ v8h load8h(const _Float16* p) { return *(const v8h*)p; }

__device__ __forceinline__ void async_fence() {
#if USE_ASYNC
#if __has_builtin(__builtin_amdgcn_s_wait_asynccnt)
  __builtin_amdgcn_s_wait_asynccnt(0);
#else
  asm volatile("s_wait_asynccnt 0" ::: "memory");
#endif
#endif
}

// Stage one tap's 64x64 f16 weight block into LDS (row pitch LPITCH).
// 512 16-byte chunks, 128 threads -> 4 chunks per lane. Uniform trip count.
__device__ __forceinline__ void stage_tap(const _Float16* __restrict__ wT,
                                          int kv, _Float16* lbuf, int tid) {
#pragma unroll
  for (int i = 0; i < 4; ++i) {
    int idx = tid + i * 128;          // 0..511
    int n  = idx >> 3;                // row (output channel)
    int ch = idx & 7;                 // 16B chunk in row
    const _Float16* g = wT + (size_t)n * KDIM + kv * C_IN + ch * 8;
    _Float16* l = lbuf + n * LPITCH + ch * 8;
#if USE_ASYNC
    __builtin_amdgcn_global_load_async_to_lds_b128(
        (g_v4i*)(uintptr_t)g,
        (l_v4i*)(unsigned)(uintptr_t)l,
        0, 0);
#else
    *(v8h*)l = load8h(g);
#endif
  }
}

// --------------------------------------------------------------------------
// Weight repack: f32 OIDHW [O][64][3][3][3] -> f16 [64][1728], k = kv*64+c.
// --------------------------------------------------------------------------
__global__ void repack_weight_kernel(const float* __restrict__ src, int O,
                                     _Float16* __restrict__ dst) {
  int i = blockIdx.x * blockDim.x + threadIdx.x;       // over 64*1728
  if (i >= 64 * KDIM) return;
  int o = i / KDIM;
  int rem = i - o * KDIM;
  int kv = rem >> 6;
  int c  = rem & 63;
  float v = (o < O) ? src[((size_t)o * C_IN + c) * KVOL + kv] : 0.f;
  dst[i] = (_Float16)v;
}

// f32 NCDHW -> f16 channel-last [b][t][h][w][c]
__global__ void x_to_f16_cl_kernel(const float* __restrict__ x,
                                   _Float16* __restrict__ xh, int total) {
  int i = blockIdx.x * blockDim.x + threadIdx.x;
  if (i >= total) return;
  int c = i & 63;
  int w = (i >> 6) & 63;
  int h = (i >> 12) & 63;
  int t = (i >> 18) & 3;
  int b = i >> 20;
  xh[i] = (_Float16)x[((((size_t)b * 64 + c) * T_D + t) * H_D + h) * W_D + w];
}

// --------------------------------------------------------------------------
// Offset conv3d (implicit GEMM): src f16 channel-last -> offf f32 [pos][64].
// --------------------------------------------------------------------------
__global__ __launch_bounds__(128) void conv_off_kernel(
    const _Float16* __restrict__ src,   // [B][T][H][W][C]
    const _Float16* __restrict__ wT,    // [64][1728]
    const float*    __restrict__ bias,  // [54]
    float*          __restrict__ offf)  // [P][64]
{
  __shared__ _Float16 lbuf[2][64 * LPITCH];   // 18 KB double buffer

  int tile = blockIdx.x;                 // 512 tiles: (b,t,h)
  int h = tile & 63;
  int t = (tile >> 6) & 3;
  int b = tile >> 8;

  int tid  = threadIdx.x;
  int wid  = tid >> 5;                   // wave id -> M tile
  int lane = tid & 31;
  int mrow = lane & 15;
  int half = lane >> 4;
  int w0 = wid << 4;

  int kt0 = (t == 0) ? 1 : 0;
  int kt1 = (t == T_D - 1) ? 2 : 3;
  int ntaps = (kt1 - kt0) * 9;

  v8f acc[4] = {};

  stage_tap(wT, kt0 * 9, lbuf[0], tid);
  async_fence();
  __syncthreads();

  for (int i = 0; i < ntaps; ++i) {
    if (i + 1 < ntaps) {
      int r1 = i + 1;
      int kvn = (kt0 + r1 / 9) * 9 + (r1 % 9);
      stage_tap(wT, kvn, lbuf[(i + 1) & 1], tid);
    }
    int kt = kt0 + i / 9;
    int r = i % 9;
    int kh = r / 3, kw = r % 3;
    int tt = t + kt - 1;                    // always in range
    int hh = h + kh - 1;
    bool hv = (unsigned)hh < H_D;
    int ww = w0 + mrow + kw - 1;
    bool v = hv && ((unsigned)ww < W_D);
    const _Float16* sp =
        src + (((size_t)(b * T_D + tt) * H_D + (hv ? hh : 0)) * W_D +
               (v ? ww : 0)) * C_IN;
    const _Float16* wl = lbuf[i & 1];
#pragma unroll
    for (int s = 0; s < 2; ++s) {
      AFrag a;
      if (v) {
        a.h[0] = load8h(sp + s * 32 + half * 8);
        a.h[1] = load8h(sp + s * 32 + half * 8 + 16);
      } else {
        a.h[0] = (v8h){};
        a.h[1] = (v8h){};
      }
      int kk = s * 32 + half * 8;
#pragma unroll
      for (int nt = 0; nt < 4; ++nt) {
        const _Float16* wp = wl + ((nt << 4) + mrow) * LPITCH + kk;
        AFrag bf;
        bf.h[0] = *(const v8h*)wp;
        bf.h[1] = *(const v8h*)(wp + 16);
        acc[nt] = __builtin_amdgcn_wmma_f32_16x16x32_f16(
            false, a.v, false, bf.v, (short)0, acc[nt], false, false);
      }
    }
    if (i + 1 < ntaps) {
      async_fence();
      __syncthreads();
    }
  }

  size_t posb = (((size_t)(b * T_D + t) * H_D + h) * W_D + w0);
#pragma unroll
  for (int nt = 0; nt < 4; ++nt) {
    int n = (nt << 4) + mrow;
    float bb = (n < OFFCH) ? bias[n] : 0.f;
#pragma unroll
    for (int r = 0; r < 8; ++r) {
      int m = r + half * 8;
      offf[(posb + m) * 64 + n] = acc[nt][r] + bb;
    }
  }
}

// --------------------------------------------------------------------------
// Deformable conv3d. MODE 0: y = leaky_relu(dcn) -> yh (f16 channel-last).
//                    MODE 1: out = dcn + x_residual (f32 NCDHW).
// --------------------------------------------------------------------------
template <int MODE>
__global__ __launch_bounds__(128) void deform_kernel(
    const _Float16* __restrict__ src,    // sampled tensor, channel-last f16
    const float*    __restrict__ offf,   // [P][64]
    const _Float16* __restrict__ wT,     // [64][1728]
    const float*    __restrict__ xres,   // residual (NCDHW, MODE 1)
    _Float16*       __restrict__ yh,     // MODE 0 output
    float*          __restrict__ outp)   // MODE 1 output
{
  __shared__ _Float16 lbuf[2][64 * LPITCH];

  int tile = blockIdx.x;
  int h = tile & 63;
  int t = (tile >> 6) & 3;
  int b = tile >> 8;

  int tid  = threadIdx.x;
  int wid  = tid >> 5;
  int lane = tid & 31;
  int mrow = lane & 15;
  int half = lane >> 4;
  int w0 = wid << 4;

  int kt0 = (t == 0) ? 1 : 0;
  int kt1 = (t == T_D - 1) ? 2 : 3;
  int ntaps = (kt1 - kt0) * 9;

  size_t posb = (((size_t)(b * T_D + t) * H_D + h) * W_D + w0);
  const float* offrow = offf + (posb + mrow) * 64;

  v8f acc[4] = {};

  stage_tap(wT, kt0 * 9, lbuf[0], tid);
  async_fence();
  __syncthreads();

  for (int i = 0; i < ntaps; ++i) {
    if (i + 1 < ntaps) {
      int r1 = i + 1;
      int kvn = (kt0 + r1 / 9) * 9 + (r1 % 9);
      stage_tap(wT, kvn, lbuf[(i + 1) & 1], tid);
    }
    int kt = kt0 + i / 9;
    int r = i % 9;
    int kh = r / 3, kw = r % 3;
    int kv = kt * 9 + r;                 // global tap id for offsets
    int tt = t + kt - 1;
    size_t tbase = (size_t)(b * T_D + tt) * H_D;

    float dh = offrow[kv * 2];
    float dw = offrow[kv * 2 + 1];
    float hs = (float)(h + kh - 1) + dh;
    float ws = (float)(w0 + mrow + kw - 1) + dw;
    float fh0 = floorf(hs), fw0 = floorf(ws);
    int h0 = (int)fh0, w0i = (int)fw0;
    int h1 = h0 + 1, w1 = w0i + 1;
    float lh = hs - fh0, lw = ws - fw0;
    float mh0 = ((unsigned)h0 < H_D) ? 1.f : 0.f;
    float mh1 = ((unsigned)h1 < H_D) ? 1.f : 0.f;
    float mw0 = ((unsigned)w0i < W_D) ? 1.f : 0.f;
    float mw1 = ((unsigned)w1 < W_D) ? 1.f : 0.f;
    float c00 = (1.f - lh) * (1.f - lw) * mh0 * mw0;
    float c01 = (1.f - lh) * lw * mh0 * mw1;
    float c10 = lh * (1.f - lw) * mh1 * mw0;
    float c11 = lh * lw * mh1 * mw1;
    int h0c = min(max(h0, 0), H_D - 1), h1c = min(max(h1, 0), H_D - 1);
    int w0c = min(max(w0i, 0), W_D - 1), w1c = min(max(w1, 0), W_D - 1);
    const _Float16* p00 = src + ((tbase + h0c) * W_D + w0c) * C_IN;
    const _Float16* p01 = src + ((tbase + h0c) * W_D + w1c) * C_IN;
    const _Float16* p10 = src + ((tbase + h1c) * W_D + w0c) * C_IN;
    const _Float16* p11 = src + ((tbase + h1c) * W_D + w1c) * C_IN;
    const _Float16* wl = lbuf[i & 1];
#pragma unroll
    for (int s = 0; s < 2; ++s) {
      AFrag a;
#pragma unroll
      for (int j = 0; j < 2; ++j) {
        int co = s * 32 + half * 8 + j * 16;
        v8f sum = __builtin_convertvector(load8h(p00 + co), v8f) * c00 +
                  __builtin_convertvector(load8h(p01 + co), v8f) * c01 +
                  __builtin_convertvector(load8h(p10 + co), v8f) * c10 +
                  __builtin_convertvector(load8h(p11 + co), v8f) * c11;
        a.h[j] = __builtin_convertvector(sum, v8h);
      }
      int kk = s * 32 + half * 8;
#pragma unroll
      for (int nt = 0; nt < 4; ++nt) {
        const _Float16* wp = wl + ((nt << 4) + mrow) * LPITCH + kk;
        AFrag bf;
        bf.h[0] = *(const v8h*)wp;
        bf.h[1] = *(const v8h*)(wp + 16);
        acc[nt] = __builtin_amdgcn_wmma_f32_16x16x32_f16(
            false, a.v, false, bf.v, (short)0, acc[nt], false, false);
      }
    }
    if (i + 1 < ntaps) {
      async_fence();
      __syncthreads();
    }
  }

#pragma unroll
  for (int nt = 0; nt < 4; ++nt) {
    int n = (nt << 4) + mrow;
#pragma unroll
    for (int r = 0; r < 8; ++r) {
      int m = r + half * 8;
      float v = acc[nt][r];
      if (MODE == 0) {
        v = (v >= 0.f) ? v : 0.1f * v;                    // leaky_relu
        yh[(posb + m) * 64 + n] = (_Float16)v;
      } else {
        size_t idx = (((size_t)(b * 64 + n) * T_D + t) * H_D + h) * W_D +
                     (w0 + m);
        outp[idx] = v + xres[idx];
      }
    }
  }
}

// --------------------------------------------------------------------------
extern "C" void kernel_launch(void* const* d_in, const int* in_sizes, int n_in,
                              void* d_out, int out_size, void* d_ws,
                              size_t ws_size, hipStream_t stream) {
  const float* x      = (const float*)d_in[0];
  const float* dcn0_w = (const float*)d_in[1];
  const float* off0_w = (const float*)d_in[2];
  const float* off0_b = (const float*)d_in[3];
  const float* dcn1_w = (const float*)d_in[4];
  const float* off1_w = (const float*)d_in[5];
  const float* off1_b = (const float*)d_in[6];
  float* out = (float*)d_out;

  char* ws = (char*)d_ws;
  const size_t WBYTES = (size_t)64 * KDIM * sizeof(_Float16);   // 221184
  _Float16* wOff0 = (_Float16*)(ws + 0 * WBYTES);
  _Float16* wDcn0 = (_Float16*)(ws + 1 * WBYTES);
  _Float16* wOff1 = (_Float16*)(ws + 2 * WBYTES);
  _Float16* wDcn1 = (_Float16*)(ws + 3 * WBYTES);
  size_t off = 4 * WBYTES;
  const size_t NPOSC = (size_t)2 * T_D * H_D * W_D * C_IN;      // 2,097,152
  _Float16* xh = (_Float16*)(ws + off);  off += NPOSC * sizeof(_Float16);
  _Float16* yh = (_Float16*)(ws + off);  off += NPOSC * sizeof(_Float16);
  float* offf  = (float*)(ws + off);     // [P][64] f32, reused for both convs

  const int WTOT = 64 * KDIM;
  repack_weight_kernel<<<(WTOT + 255) / 256, 256, 0, stream>>>(off0_w, OFFCH, wOff0);
  repack_weight_kernel<<<(WTOT + 255) / 256, 256, 0, stream>>>(dcn0_w, 64,    wDcn0);
  repack_weight_kernel<<<(WTOT + 255) / 256, 256, 0, stream>>>(off1_w, OFFCH, wOff1);
  repack_weight_kernel<<<(WTOT + 255) / 256, 256, 0, stream>>>(dcn1_w, 64,    wDcn1);

  const int XTOT = (int)NPOSC;
  x_to_f16_cl_kernel<<<(XTOT + 255) / 256, 256, 0, stream>>>(x, xh, XTOT);

  const int NROWS = 2 * T_D * H_D;                              // 512 blocks
  conv_off_kernel<<<NROWS, 128, 0, stream>>>(xh, wOff0, off0_b, offf);
  deform_kernel<0><<<NROWS, 128, 0, stream>>>(xh, offf, wDcn0, nullptr, yh, nullptr);
  conv_off_kernel<<<NROWS, 128, 0, stream>>>(yh, wOff1, off1_b, offf);
  deform_kernel<1><<<NROWS, 128, 0, stream>>>(yh, offf, wDcn1, x, nullptr, out);
}
